// DAGBF_87445534147213
// MI455X (gfx1250) — compile-verified, matrix operand order checked by reading, not run on hardware
//
#include <hip/hip_runtime.h>
#include <math.h>

// ---------------------------------------------------------------------------
// Problem constants (from reference): B=2, Cin=4, H=W=256, HID=16, K=25
// ---------------------------------------------------------------------------
#define HID   16
#define KPTS  25
#define H_    256
#define W_    256
#define PW_NOUT     75
#define PRM_STRIDE  80

typedef __attribute__((ext_vector_type(16))) _Float16 v16h;
typedef __attribute__((ext_vector_type(8)))  float    v8f;

// ---------------------------------------------------------------------------
// Helpers
// ---------------------------------------------------------------------------
__device__ __forceinline__ float gelu_exact(float v) {
    return 0.5f * v * (1.0f + erff(v * 0.70710678118654752f));
}

__device__ __forceinline__ float reflectf(float c, int size) {
    float m = 2.0f * (float)(size - 1);
    c = fmodf(fabsf(c), m);
    return (c > (float)(size - 1)) ? (m - c) : c;
}

__device__ __forceinline__ int reflecti(int i, int n) {
    if (i < 0)  i = -i;
    if (i >= n) i = 2 * n - 2 - i;
    return i;
}

// A fragment: 16 rows (pixels) x 16 K (channels), K padded 16->32 with zeros.
// ISA 7.12.2 f16 A 16x32 layout: lane&15 = M, (lane>>4)*8 = K half-offset,
// halves e=0..7 -> K=khalf+e, e=8..15 -> K=16+khalf+(e-8) (zeroed here).
__device__ __forceinline__ v16h load_A_frag(const float* rows, int row_stride) {
    int lane  = threadIdx.x & 31;
    int m     = lane & 15;
    int khalf = (lane >> 4) * 8;
    const float* rp = rows + (size_t)m * row_stride + khalf;
    v16h a;
#pragma unroll
    for (int e = 0; e < 8; ++e)  a[e] = (_Float16)rp[e];
#pragma unroll
    for (int e = 8; e < 16; ++e) a[e] = (_Float16)0.0f;
    return a;
}

// B fragment from row-major weight (nout, 16): B[k][n] = w[n][k].
// lane&15 = N column, (lane>>4)*8 = K half-offset; K>=16 zero.
__device__ __forceinline__ v16h load_B_frag(const float* w, int nout, int ntile) {
    int lane  = threadIdx.x & 31;
    int n     = ntile * 16 + (lane & 15);
    int khalf = (lane >> 4) * 8;
    v16h b;
#pragma unroll
    for (int e = 0; e < 16; ++e) b[e] = (_Float16)0.0f;
    if (n < nout) {
        const float* rp = w + (size_t)n * HID + khalf;
#pragma unroll
        for (int e = 0; e < 8; ++e) b[e] = (_Float16)rp[e];
    }
    return b;
}

// ---------------------------------------------------------------------------
// K1: input projections + GELU + RMSNorm.  x:(B,4,H,W) channel-major.
// Outputs pixel-major [p][16]: shortcut, feat0, xn1.
// ---------------------------------------------------------------------------
__global__ void k1_proj(const float* __restrict__ x,
                        const float* __restrict__ pm_w, const float* __restrict__ pm_b,
                        const float* __restrict__ pa_w, const float* __restrict__ pa_b,
                        const float* __restrict__ n1w,
                        float* __restrict__ shortcut, float* __restrict__ feat0,
                        float* __restrict__ xn1, int npix) {
    int p = blockIdx.x * blockDim.x + threadIdx.x;
    if (p >= npix) return;
    int hw = H_ * W_;
    int b = p / hw, rem = p - b * hw;
    const float* xb = x + (size_t)b * 4 * hw + rem;
    float x0 = xb[0], x1 = xb[hw], x2 = xb[2 * hw], x3 = xb[3 * hw];
    float f[HID];
    float ss = 0.f;
#pragma unroll
    for (int o = 0; o < HID; ++o) {
        float s = pm_w[o] * x0 + pm_b[o]
                + pa_w[o * 3 + 0] * x1 + pa_w[o * 3 + 1] * x2 + pa_w[o * 3 + 2] * x3
                + pa_b[o];
        shortcut[(size_t)p * HID + o] = s;
        float g = gelu_exact(s);
        feat0[(size_t)p * HID + o] = g;
        f[o] = g;
        ss += g * g;
    }
    float rms = sqrtf(ss) * 0.25f;           // sqrt(sum)/sqrt(16)
    float inv = 1.0f / (rms + 1e-6f);
#pragma unroll
    for (int o = 0; o < HID; ++o)
        xn1[(size_t)p * HID + o] = f[o] * inv * n1w[o];
}

// ---------------------------------------------------------------------------
// K2: depthwise 3x3 conv (reflect boundary) + GELU, pixel-major in/out.
// ---------------------------------------------------------------------------
__global__ void k2_dwgelu(const float* __restrict__ src,
                          const float* __restrict__ w9, const float* __restrict__ bias,
                          float* __restrict__ dst, int npix) {
    int p = blockIdx.x * blockDim.x + threadIdx.x;
    if (p >= npix) return;
    int hw = H_ * W_;
    int b = p / hw, rem = p - b * hw;
    int y = rem / W_, xx = rem - y * W_;
    float acc[HID];
#pragma unroll
    for (int o = 0; o < HID; ++o) acc[o] = bias[o];
#pragma unroll
    for (int ky = 0; ky < 3; ++ky) {
        int yy = reflecti(y + ky - 1, H_);
#pragma unroll
        for (int kx = 0; kx < 3; ++kx) {
            int xr = reflecti(xx + kx - 1, W_);
            const float* row = src + ((size_t)b * hw + (size_t)yy * W_ + xr) * HID;
#pragma unroll
            for (int o = 0; o < HID; ++o)
                acc[o] += w9[o * 9 + ky * 3 + kx] * row[o];
        }
    }
#pragma unroll
    for (int o = 0; o < HID; ++o)
        dst[(size_t)p * HID + o] = gelu_exact(acc[o]);
}

// ---------------------------------------------------------------------------
// K3: WMMA pointwise conv 16->75, then per-pixel softmax + deformable sample.
// Block = 128 threads = 4 waves; each wave computes 32 pixels (2 M-tiles x
// 5 N-tiles of v_wmma_f32_16x16x32_f16).  Params staged through LDS.
// ---------------------------------------------------------------------------
__global__ void __launch_bounds__(128)
k3_pw_sample(const float* __restrict__ h, const float* __restrict__ pw_w,
             const float* __restrict__ pw_b, const float* __restrict__ src,
             float* __restrict__ out, float spacing, float max_off, int npix) {
    __shared__ float prm[128 * PRM_STRIDE];          // 40 KB
    int blockbase = blockIdx.x * 128;
    int lane = threadIdx.x & 31;
    int wv   = threadIdx.x >> 5;

    if (lane == 0) __builtin_prefetch(pw_w, 0, 0);   // global_prefetch_b8

    // ---- Phase 1: GEMM [32px x 16] * [16 x 80(pad)] per wave via WMMA ----
#pragma unroll
    for (int t = 0; t < 2; ++t) {
        int mloc = wv * 32 + t * 16;
        v16h a = load_A_frag(h + (size_t)(blockbase + mloc) * HID, HID);
#pragma unroll
        for (int nt = 0; nt < 5; ++nt) {
            v16h bf = load_B_frag(pw_w, PW_NOUT, nt);
            v8f  c  = {};
            c = __builtin_amdgcn_wmma_f32_16x16x32_f16(false, a, false, bf,
                                                       (short)0, c, false, false);
            int n     = nt * 16 + (lane & 15);
            int mhalf = (lane >> 4) * 8;
            if (n < PW_NOUT) {
                float bb = pw_b[n];
#pragma unroll
                for (int r = 0; r < 8; ++r)
                    prm[(mloc + mhalf + r) * PRM_STRIDE + n] = c[r] + bb;
            }
        }
    }
    __syncthreads();

    // ---- Phase 2: per-pixel softmax over 25 logits + bilinear gather ----
    int p = blockbase + threadIdx.x;
    if (p >= npix) return;
    const float* myp = prm + threadIdx.x * PRM_STRIDE;

    float mx = -1e30f;
#pragma unroll
    for (int k = 0; k < KPTS; ++k) mx = fmaxf(mx, myp[50 + k]);
    float wk[KPTS], sum = 0.f;
#pragma unroll
    for (int k = 0; k < KPTS; ++k) { wk[k] = expf(myp[50 + k] - mx); sum += wk[k]; }
    float invs = 1.0f / sum;

    int hw = H_ * W_;
    int b = p / hw, rem = p - b * hw;
    int y = rem / W_, xx = rem - y * W_;
    const float* srcb = src + (size_t)b * hw * HID;

    float acc[HID];
#pragma unroll
    for (int o = 0; o < HID; ++o) acc[o] = 0.f;

#pragma unroll 1
    for (int k = 0; k < KPTS; ++k) {
        float bx = (float)((k % 5) - 2) * spacing;
        float by = (float)((k / 5) - 2) * spacing;
        float ix = (float)xx + bx + tanhf(myp[2 * k])     * max_off;
        float iy = (float)y  + by + tanhf(myp[2 * k + 1]) * max_off;
        ix = reflectf(ix, W_); iy = reflectf(iy, H_);
        float x0f = floorf(ix), y0f = floorf(iy);
        float wx = ix - x0f, wy = iy - y0f;
        int x0 = min(max((int)x0f, 0), W_ - 1); int x1 = min(x0 + 1, W_ - 1);
        int y0 = min(max((int)y0f, 0), H_ - 1); int y1 = min(y0 + 1, H_ - 1);
        const float* p00 = srcb + ((size_t)y0 * W_ + x0) * HID;
        const float* p01 = srcb + ((size_t)y0 * W_ + x1) * HID;
        const float* p10 = srcb + ((size_t)y1 * W_ + x0) * HID;
        const float* p11 = srcb + ((size_t)y1 * W_ + x1) * HID;
        float g   = wk[k] * invs;
        float w00 = (1.f - wx) * (1.f - wy) * g;
        float w01 = wx * (1.f - wy) * g;
        float w10 = (1.f - wx) * wy * g;
        float w11 = wx * wy * g;
#pragma unroll
        for (int o = 0; o < HID; ++o)
            acc[o] += w00 * p00[o] + w01 * p01[o] + w10 * p10[o] + w11 * p11[o];
    }
    float* op = out + (size_t)p * HID;
#pragma unroll
    for (int o = 0; o < HID; ++o) op[o] = acc[o];
}

// ---------------------------------------------------------------------------
// K6: rmsnorm(a2) -> gate (WMMA 16x16) -> *feat0 -> proj_out (WMMA) + shortcut
// ---------------------------------------------------------------------------
__global__ void __launch_bounds__(128)
k6_gate_proj(const float* __restrict__ a2, const float* __restrict__ feat0,
             const float* __restrict__ shortcut, const float* __restrict__ n2w,
             const float* __restrict__ gate_w, const float* __restrict__ gate_b,
             const float* __restrict__ po_w,  const float* __restrict__ po_b,
             float* __restrict__ feat, int npix) {
    __shared__ float sb[128 * HID];                  // 8 KB
    int blockbase = blockIdx.x * 128;
    int p = blockbase + threadIdx.x;
    (void)npix;

    // Phase A: rmsnorm into LDS (per thread = per pixel)
    {
        const float* ap = a2 + (size_t)p * HID;
        float v[HID], ss = 0.f;
#pragma unroll
        for (int o = 0; o < HID; ++o) { v[o] = ap[o]; ss += v[o] * v[o]; }
        float inv = 1.0f / (sqrtf(ss) * 0.25f + 1e-6f);
#pragma unroll
        for (int o = 0; o < HID; ++o)
            sb[threadIdx.x * HID + o] = v[o] * inv * n2w[o];
    }
    __syncthreads();

    int lane  = threadIdx.x & 31;
    int wv    = threadIdx.x >> 5;
    int nloc  = lane & 15;
    int mhalf = (lane >> 4) * 8;

    // Phase B: gate conv; multiply by feat0; write back to LDS (own rows only)
#pragma unroll
    for (int t = 0; t < 2; ++t) {
        int mloc = wv * 32 + t * 16;
        v16h aA = load_A_frag(sb + mloc * HID, HID);
        v16h bB = load_B_frag(gate_w, HID, 0);
        v8f  c  = {};
        c = __builtin_amdgcn_wmma_f32_16x16x32_f16(false, aA, false, bB,
                                                   (short)0, c, false, false);
#pragma unroll
        for (int r = 0; r < 8; ++r) {
            int m = mloc + mhalf + r;
            float g = c[r] + gate_b[nloc];
            sb[m * HID + nloc] = feat0[((size_t)blockbase + m) * HID + nloc] * g;
        }
    }
    __syncthreads();

    // Phase C: proj_out + shortcut -> feat (global)
#pragma unroll
    for (int t = 0; t < 2; ++t) {
        int mloc = wv * 32 + t * 16;
        v16h aA = load_A_frag(sb + mloc * HID, HID);
        v16h bB = load_B_frag(po_w, HID, 0);
        v8f  c  = {};
        c = __builtin_amdgcn_wmma_f32_16x16x32_f16(false, aA, false, bB,
                                                   (short)0, c, false, false);
#pragma unroll
        for (int r = 0; r < 8; ++r) {
            int m = mloc + mhalf + r;
            size_t gi = ((size_t)blockbase + m) * HID + nloc;
            feat[gi] = c[r] + po_b[nloc] + shortcut[gi];
        }
    }
}

// ---------------------------------------------------------------------------
// K7: heads (sigma / offset) + bilateral combination -> output (B,1,H,W)
// ---------------------------------------------------------------------------
__global__ void k7_final(const float* __restrict__ feat, const float* __restrict__ x,
                         const float* __restrict__ sw, const float* __restrict__ sbias,
                         const float* __restrict__ ow, const float* __restrict__ ob,
                         float* __restrict__ out, int npix) {
    int p = blockIdx.x * blockDim.x + threadIdx.x;
    if (p >= npix) return;
    int hw = H_ * W_;
    int b = p / hw, rem = p - b * hw;
    int y = rem / W_, xx = rem - y * W_;

    const float* fp = feat + (size_t)p * HID;
    float f[HID];
#pragma unroll
    for (int o = 0; o < HID; ++o) f[o] = fp[o];

    // sigma head: softplus, clamp to 6, + eps
    float sig[3];
#pragma unroll
    for (int j = 0; j < 3; ++j) {
        float s = sbias[j];
#pragma unroll
        for (int c = 0; c < HID; ++c) s += sw[j * HID + c] * f[c];
        float sp = (s > 20.f) ? s : log1pf(expf(s));
        sig[j] = fminf(sp, 6.0f) + 1e-6f;
    }
    float sx2 = 2.f * sig[0] * sig[0];
    float sy2 = 2.f * sig[1] * sig[1];
    float sr2 = 2.f * sig[2] * sig[2];

    const float* x0p = x + (size_t)b * 4 * hw;       // channel 0 of input
    const float* fb  = feat + (size_t)b * hw * HID;

    float num = 0.f, den = 0.f;
#pragma unroll 1
    for (int k = 0; k < KPTS; ++k) {
        float d0 = ob[2 * k], d1 = ob[2 * k + 1];
#pragma unroll
        for (int c = 0; c < HID; ++c) {
            d0 += ow[(2 * k)     * HID + c] * f[c];
            d1 += ow[(2 * k + 1) * HID + c] * f[c];
        }
        float dx = (float)((k % 5) - 2) + tanhf(d0) * 5.0f;
        float dy = (float)((k / 5) - 2) + tanhf(d1) * 5.0f;
        float ix = reflectf((float)xx + dx, W_);
        float iy = reflectf((float)y  + dy, H_);
        float x0f = floorf(ix), y0f = floorf(iy);
        float wxf = ix - x0f, wyf = iy - y0f;
        int x0 = min(max((int)x0f, 0), W_ - 1); int x1 = min(x0 + 1, W_ - 1);
        int y0 = min(max((int)y0f, 0), H_ - 1); int y1 = min(y0 + 1, H_ - 1);
        float w00 = (1.f - wxf) * (1.f - wyf), w01 = wxf * (1.f - wyf);
        float w10 = (1.f - wxf) * wyf,         w11 = wxf * wyf;

        float patch = w00 * x0p[y0 * W_ + x0] + w01 * x0p[y0 * W_ + x1]
                    + w10 * x0p[y1 * W_ + x0] + w11 * x0p[y1 * W_ + x1];

        const float* p00 = fb + ((size_t)y0 * W_ + x0) * HID;
        const float* p01 = fb + ((size_t)y0 * W_ + x1) * HID;
        const float* p10 = fb + ((size_t)y1 * W_ + x0) * HID;
        const float* p11 = fb + ((size_t)y1 * W_ + x1) * HID;
        float fd = 0.f;
#pragma unroll
        for (int c = 0; c < HID; ++c) {
            float s  = w00 * p00[c] + w01 * p01[c] + w10 * p10[c] + w11 * p11[c];
            float df = f[c] - s;
            fd += df * df;
        }
        float spat = expf(-(dx * dx / sx2 + dy * dy / sy2));
        float rk   = expf(-fd / sr2);
        float wgt  = spat * rk;
        num += patch * wgt;
        den += wgt;
    }
    out[p] = num / (den + 1e-8f);
}

// ---------------------------------------------------------------------------
// Launch
// ---------------------------------------------------------------------------
extern "C" void kernel_launch(void* const* d_in, const int* in_sizes, int n_in,
                              void* d_out, int out_size, void* d_ws, size_t ws_size,
                              hipStream_t stream) {
    const float* x      = (const float*)d_in[0];
    const float* pm_w   = (const float*)d_in[1];
    const float* pm_b   = (const float*)d_in[2];
    const float* pa_w   = (const float*)d_in[3];
    const float* pa_b   = (const float*)d_in[4];
    const float* n1w    = (const float*)d_in[5];
    const float* f_dw_w = (const float*)d_in[6];
    const float* f_dw_b = (const float*)d_in[7];
    const float* f_pw_w = (const float*)d_in[8];
    const float* f_pw_b = (const float*)d_in[9];
    const float* c_dw_w = (const float*)d_in[10];
    const float* c_dw_b = (const float*)d_in[11];
    const float* c_pw_w = (const float*)d_in[12];
    const float* c_pw_b = (const float*)d_in[13];
    const float* n2w    = (const float*)d_in[14];
    const float* gate_w = (const float*)d_in[15];
    const float* gate_b = (const float*)d_in[16];
    const float* po_w   = (const float*)d_in[17];
    const float* po_b   = (const float*)d_in[18];
    const float* sig_w  = (const float*)d_in[19];
    const float* sig_b  = (const float*)d_in[20];
    const float* oh_w   = (const float*)d_in[21];
    const float* oh_b   = (const float*)d_in[22];
    (void)n_in; (void)ws_size; (void)out_size;

    int npix = in_sizes[0] / 4;                  // B*H*W
    size_t S = (size_t)npix * HID;
    float* ws       = (float*)d_ws;
    float* shortcut = ws;
    float* feat0    = ws + S;
    float* xn1      = ws + 2 * S;                // reused as a2 later
    float* hbuf     = ws + 3 * S;
    float* a1       = ws + 4 * S;
    float* feat     = ws + 5 * S;

    int tb = 256;
    int gb = (npix + tb - 1) / tb;
    int gw = (npix + 127) / 128;

    k1_proj    <<<gb, tb, 0, stream>>>(x, pm_w, pm_b, pa_w, pa_b, n1w,
                                       shortcut, feat0, xn1, npix);
    // fine deformable attention
    k2_dwgelu  <<<gb, tb, 0, stream>>>(xn1, f_dw_w, f_dw_b, hbuf, npix);
    k3_pw_sample<<<gw, 128, 0, stream>>>(hbuf, f_pw_w, f_pw_b, xn1, a1,
                                         1.0f, 4.0f, npix);
    // coarse deformable attention (a2 -> xn1 buffer)
    k2_dwgelu  <<<gb, tb, 0, stream>>>(a1, c_dw_w, c_dw_b, hbuf, npix);
    k3_pw_sample<<<gw, 128, 0, stream>>>(hbuf, c_pw_w, c_pw_b, a1, xn1,
                                         3.0f, 6.0f, npix);
    // gate + proj_out + shortcut
    k6_gate_proj<<<gw, 128, 0, stream>>>(xn1, feat0, shortcut, n2w,
                                         gate_w, gate_b, po_w, po_b, feat, npix);
    // heads + bilateral combine
    k7_final   <<<gb, tb, 0, stream>>>(feat, x, sig_w, sig_b, oh_w, oh_b,
                                       (float*)d_out, npix);
}
